// IWLayer_58085137711115
// MI455X (gfx1250) — matrix operand order repaired and checked by reading, not compiled, not run on hardware
//
#include <hip/hip_runtime.h>
#include <hip/hip_bf16.h>
#include <math.h>

// ---------------- problem constants ----------------
#define S_DIM 64
#define P_DIM 128
#define PT_DIM 2048
#define PP (P_DIM * P_DIM)      // 16384
#define PPT (P_DIM * PT_DIM)    // 262144
#define LDP 129                 // LDS padded stride (64 banks, avoid stride-128 conflicts)

typedef __attribute__((ext_vector_type(16))) _Float16 v16h;
typedef __attribute__((ext_vector_type(8)))  float    v8f;
typedef __attribute__((ext_vector_type(4)))  unsigned v4u;
typedef __attribute__((ext_vector_type(8)))  unsigned v8u;

// ---------------- deterministic hash RNG (fast-math transcendentals) ----------------
__device__ __forceinline__ unsigned pcg(unsigned v) {
    v = v * 747796405u + 2891336453u;
    unsigned w = ((v >> ((v >> 28) + 4u)) ^ v) * 277803737u;
    return (w >> 22) ^ w;
}
__device__ __forceinline__ float u01f(unsigned h) {
    return ((float)(h >> 8) + 0.5f) * (1.0f / 16777216.0f);
}
__device__ __forceinline__ float rng_normal(unsigned s) {
    float u1 = u01f(pcg(s));
    float u2 = u01f(pcg(s ^ 0x9E3779B9u));
    return sqrtf(-2.0f * __logf(u1)) * __cosf(6.28318530718f * u2);
}
// Marsaglia-Tsang, alpha >= 1 (here alpha in [1.5, ~1089])
__device__ __forceinline__ float rng_gamma(float alpha, unsigned s) {
    float d = alpha - (1.0f / 3.0f);
    float c = rsqrtf(9.0f * d);
#pragma unroll 1
    for (int it = 0; it < 12; ++it) {
        unsigned s2 = s + 0x632BE5ABu * (unsigned)it;
        float x = rng_normal(s2);
        float v = 1.0f + c * x;
        if (v <= 0.0f) continue;
        v = v * v * v;
        float u = u01f(pcg(s2 ^ 0x85EBCA6Bu));
        if (__logf(u) < 0.5f * x * x + d - d * v + d * __logf(v)) return d * v;
    }
    return d;  // fallback ~mean
}

// ---------------- W = diag * V V^T / P ----------------
__global__ void k_vdv(const float* __restrict__ V, const float* __restrict__ log_diag,
                      float* __restrict__ W) {
    int i = blockIdx.x, j = threadIdx.x;
    float dg = __expf(log_diag[0]);
    float acc = 0.0f;
    for (int k = 0; k < P_DIM; ++k) acc += V[i * P_DIM + k] * V[j * P_DIM + k];
    W[i * P_DIM + j] = acc * dg * (1.0f / (float)P_DIM);
}

// ---------------- psi = delta*Kii + W ; dKii = delta*Kii ----------------
__global__ void k_prep(const float* __restrict__ Kii, const float* __restrict__ W,
                       const float* __restrict__ log_delta,
                       float* __restrict__ psi, float* __restrict__ dKii) {
    long long idx = (long long)blockIdx.x * blockDim.x + threadIdx.x;  // S*PP
    float delta = __expf(log_delta[0]);
    float dk = delta * Kii[idx];
    dKii[idx] = dk;
    psi[idx] = dk + W[idx & (PP - 1)];
}

// ---------------- in-place batched Cholesky (lower), one block per sample ----------------
__global__ void k_chol(float* __restrict__ Mat) {
    extern __shared__ float T[];  // P_DIM * LDP
    float* Ms = Mat + (long long)blockIdx.x * PP;
    int t = threadIdx.x;  // 128 threads: thread t owns row t
    for (int r = 0; r < P_DIM; ++r) T[r * LDP + t] = Ms[r * P_DIM + t];
    __syncthreads();
    for (int j = 0; j < P_DIM; ++j) {
        float piv = T[j * LDP + j];
        __syncthreads();
        float sj = sqrtf(piv);
        if (t > j) T[t * LDP + j] = T[t * LDP + j] / sj;
        if (t == j) T[j * LDP + j] = sj;
        __syncthreads();
        if (t > j) {
            float lij = T[t * LDP + j];
            for (int k = j + 1; k <= t; ++k) T[t * LDP + k] -= lij * T[k * LDP + j];
        }
        __syncthreads();
    }
    for (int r = 0; r < P_DIM; ++r) Ms[r * P_DIM + t] = (t <= r) ? T[r * LDP + t] : 0.0f;
}

// ---------------- Bartlett sample A (lower-tri) ----------------
__global__ void k_bartlett(float* __restrict__ A, const float* __restrict__ log_delta,
                           const float* __restrict__ log_gamma) {
    long long idx = (long long)blockIdx.x * blockDim.x + threadIdx.x;  // S*PP
    int ij = (int)(idx & (PP - 1));
    int i = ij >> 7, j = ij & (P_DIM - 1);
    float out;
    if (j > i) {
        out = 0.0f;
    } else if (j < i) {
        out = rng_normal(0xB5297A4Du ^ (unsigned)idx * 0x9E3779B1u);
    } else {
        float delta = __expf(log_delta[0]);
        float gam = __expf(log_gamma[0]);
        float post_nu = delta + (float)P_DIM + 1.0f + gam;
        float dof = post_nu - (float)i;
        out = sqrtf(2.0f * rng_gamma(0.5f * dof, 0x1B873593u ^ (unsigned)idx * 0x85EBCA77u));
    }
    A[idx] = out;
}

// ---------------- lower-triangular inverse, one block per sample ----------------
__global__ void k_trinv(const float* __restrict__ Lmat, float* __restrict__ Out) {
    extern __shared__ float sh[];  // 2 * P_DIM * LDP
    float* Lsh = sh;
    float* Xsh = sh + P_DIM * LDP;
    long long b = blockIdx.x;
    const float* src = Lmat + b * PP;
    float* dst = Out + b * PP;
    int c = threadIdx.x;  // thread c owns column c of the inverse
    for (int r = 0; r < P_DIM; ++r) {
        Lsh[r * LDP + c] = src[r * P_DIM + c];
        Xsh[r * LDP + c] = 0.0f;
    }
    __syncthreads();
    for (int i = 0; i < P_DIM; ++i) {
        if (c <= i) {
            float acc = (i == c) ? 1.0f : 0.0f;
            for (int j = c; j < i; ++j) acc -= Lsh[i * LDP + j] * Xsh[j * LDP + c];
            Xsh[i * LDP + c] = acc / Lsh[i * LDP + i];
        }
        __syncthreads();
    }
    for (int r = 0; r < P_DIM; ++r) dst[r * P_DIM + c] = Xsh[r * LDP + c];
}

// ---------------- fragment index helpers (ISA 7.12.2 f16 layouts) ----------------
__device__ __forceinline__ int a_kidx(int h, int g) {
    int v = h >> 1, r = h & 1;
    return ((v < 4) ? 0 : 16) + g * 8 + (v & 3) * 2 + r;
}

// ---------------- small batched WMMA GEMM (128x128x128): one wave per 16x16 tile ----
__global__ void k_gemm_small(const float* __restrict__ Ab, const float* __restrict__ Bb,
                             float* __restrict__ Db,
                             long long sA, long long sB, long long sD,
                             int ldaR, int ldaC, int ldbR, int ldbC, int ldd) {
    const int lane = threadIdx.x & 31;
    const int g = lane >> 4;
    const int mn = lane & 15;
    const long long b = blockIdx.z;
    const float* A = Ab + b * sA;
    const float* B = Bb + b * sB;
    float* D = Db + b * sD;
    const int m0 = blockIdx.y * 16, n0 = blockIdx.x * 16;

    v8f c = {};
#pragma unroll
    for (int k0 = 0; k0 < P_DIM; k0 += 32) {
        v16h af, bf;
#pragma unroll
        for (int h = 0; h < 16; ++h) {
            int ka = g * 16 + h;
            af[h] = (_Float16)A[(long long)(m0 + mn) * ldaR + (long long)(k0 + ka) * ldaC];
            int kb = g * 16 + h;
            bf[h] = (_Float16)B[(long long)(k0 + kb) * ldbR + (long long)(n0 + mn) * ldbC];
        }
        c = __builtin_amdgcn_wmma_f32_16x16x32_f16(false, af, false, bf, (short)0, c,
                                                   false, false);
    }
#pragma unroll
    for (int i = 0; i < 8; ++i)
        D[(long long)(m0 + i + 8 * g) * ldd + (n0 + mn)] = c[i];
}

// ---------------- big batched WMMA GEMM (128 x PT x 128) -----------------------------
// Block = 256 threads = 8 waves; each block owns a 128x16 strip of D.
// The 128x128 A panel is DMA'd to LDS once per block by the Tensor Data Mover
// (tensor_load_to_lds, 2-D descriptor, TENSORcnt), then all 8 waves read their
// A fragments from LDS while B streams coalesced from global (lanes span N).
// GENB: B entries are N(0,1) hashed from (batch,k,n) (eps generated in-loader).
template <bool GENB>
__global__ void k_gemm_big(const float* __restrict__ Ab, const float* __restrict__ Bb,
                           float* __restrict__ Db,
                           long long sA, long long sB, long long sD,
                           int ldaR, int ldaC, int ldbR, int ldbC, int ldd,
                           const float* scaleBlog, unsigned seed) {
    extern __shared__ float shA[];  // 128*128 f32 = 64KB
    const long long b = blockIdx.z;
    const float* Ag = Ab + b * sA;
    const int wave = threadIdx.x >> 5;
    const int lane = threadIdx.x & 31;
    const int g = lane >> 4, mn = lane & 15;

    // --- TDM: one async tensor DMA of the A panel into LDS (wave 0 issues) ---
    if (wave == 0) {
        unsigned long long ga = (unsigned long long)(const void*)Ag;
        v4u g0;
        v8u g1;
        g0[0] = 1u;                                   // count=1 (valid user D#)
        g0[1] = 0u;                                   // lds_addr = 0 (dynamic LDS base)
        g0[2] = (unsigned)(ga & 0xFFFFFFFFu);         // global_addr[31:0]
        g0[3] = (unsigned)((ga >> 32) & 0x01FFFFFFu)  // global_addr[56:32]
                | (2u << 30);                         // type = 2 ("image")
        g1[0] = (2u << 16);                           // data_size=4B, mask=0, no pad/iter
        g1[1] = (P_DIM & 0xFFFFu) << 16;              // tensor_dim0[15:0] = 128
        g1[2] = ((P_DIM >> 16) & 0xFFFFu) | ((P_DIM & 0xFFFFu) << 16);  // dim0 hi | dim1 lo
        g1[3] = ((P_DIM >> 16) & 0xFFFFu) | ((unsigned)P_DIM << 16);    // dim1 hi | tile_dim0
        g1[4] = (unsigned)P_DIM;                      // tile_dim1=128, tile_dim2=0
        g1[5] = (unsigned)P_DIM;                      // tensor_dim0_stride[31:0] = 128
        g1[6] = 0u;                                   // stride0 hi | stride1 lo
        g1[7] = 0u;                                   // stride1 hi
        asm volatile("tensor_load_to_lds %0, %1" ::"s"(g0), "s"(g1) : "memory");
        __builtin_amdgcn_s_wait_tensorcnt(0);
    }
    __syncthreads();

    const float* Bg = Bb + b * sB;
    float* D = Db + b * sD;
    const int m0 = wave * 16;             // 8 waves cover M = 0..127
    const int n0 = blockIdx.x * 16;
    const float sBv = scaleBlog ? __expf(scaleBlog[0]) : 1.0f;

    v8f c = {};
#pragma unroll
    for (int k0 = 0; k0 < P_DIM; k0 += 32) {
        if (!GENB && k0 + 32 < P_DIM)  // prefetch next B k-strip (global_prefetch_b8)
            __builtin_prefetch(&Bg[(long long)(k0 + 32) * ldbR + (long long)(n0 + mn) * ldbC],
                               0, 1);
        v16h af, bf;
#pragma unroll
        for (int h = 0; h < 16; ++h) {
            int ka = g * 16 + h;
            af[h] = (_Float16)shA[(m0 + mn) * ldaR + (k0 + ka) * ldaC];
            int kb = g * 16 + h;
            float bv;
            if (GENB) {
                unsigned id = (unsigned)((b * P_DIM + (k0 + kb)) * (long long)PT_DIM +
                                         (n0 + mn));
                bv = rng_normal(seed ^ (id * 0x01000193u));
            } else {
                bv = Bg[(long long)(k0 + kb) * ldbR + (long long)(n0 + mn) * ldbC];
            }
            bf[h] = (_Float16)(bv * sBv);
        }
        c = __builtin_amdgcn_wmma_f32_16x16x32_f16(false, af, false, bf, (short)0, c,
                                                   false, false);
    }
#pragma unroll
    for (int i = 0; i < 8; ++i)
        D[(long long)(m0 + i + 8 * g) * ldd + (n0 + mn)] = c[i];
}

// ---------------- gtti = dktti / (2*Gamma(alpha)) ----------------
__global__ void k_gtti(const float* __restrict__ Kit, const float* __restrict__ Ktt,
                       const float* __restrict__ X0, const float* __restrict__ log_delta,
                       float* __restrict__ gtti) {
    long long idx = (long long)blockIdx.x * blockDim.x + threadIdx.x;  // S*PT
    int s = (int)(idx >> 11);
    int t = (int)(idx & (PT_DIM - 1));
    float delta = __expf(log_delta[0]);
    const float* kit = Kit + (long long)s * PPT;
    const float* x0 = X0 + (long long)s * PPT;
    float acc = 0.0f;
    for (int p = 0; p < P_DIM; ++p)
        acc += kit[(long long)p * PT_DIM + t] * x0[(long long)p * PT_DIM + t];
    float dktti = delta * Ktt[idx] - delta * acc;  // dkit . invKii_kit, dkit = delta*Kit
    float alpha = 0.5f * (delta + (float)P_DIM + (float)PT_DIM + 1.0f);
    float gs = rng_gamma(alpha, 0xC2B2AE35u ^ (unsigned)idx * 0x27D4EB2Fu);
    gtti[idx] = dktti / (2.0f * gs);
}

// ---------------- X = invKii_kit + noise * sqrt(gtti) (in place on X0) ----------------
__global__ void k_form_X(float* __restrict__ X0, const float* __restrict__ noise,
                         const float* __restrict__ gtti) {
    long long idx = (long long)blockIdx.x * blockDim.x + threadIdx.x;  // S*PPT
    long long s = idx >> 18;  // PPT = 2^18
    int t = (int)(idx & (PT_DIM - 1));
    float sg = sqrtf(fmaxf(gtti[s * PT_DIM + t], 0.0f));
    X0[idx] = X0[idx] + noise[idx] * sg;
}

// ---------------- gtt = gtti + sum_p git*X ----------------
__global__ void k_gtt(const float* __restrict__ git, const float* __restrict__ X,
                      const float* __restrict__ gtti, float* __restrict__ gtt) {
    long long idx = (long long)blockIdx.x * blockDim.x + threadIdx.x;  // S*PT
    int s = (int)(idx >> 11);
    int t = (int)(idx & (PT_DIM - 1));
    const float* gs = git + (long long)s * PPT;
    const float* xs = X + (long long)s * PPT;
    float acc = 0.0f;
    for (int p = 0; p < P_DIM; ++p)
        acc += gs[(long long)p * PT_DIM + t] * xs[(long long)p * PT_DIM + t];
    gtt[idx] = gtti[idx] + acc;
}

// ---------------- per-sample log p - log q ----------------
__device__ __forceinline__ float bsum(float v, float* red) {
    int t = threadIdx.x;
    red[t] = v;
    __syncthreads();
    for (int o = 64; o > 0; o >>= 1) {
        if (t < o) red[t] += red[t + o];
        __syncthreads();
    }
    float r = red[0];
    __syncthreads();
    return r;
}

__global__ void k_logpq(const float* __restrict__ L, const float* __restrict__ Lk,
                        const float* __restrict__ A, const float* __restrict__ MA,
                        const float* __restrict__ log_delta,
                        const float* __restrict__ log_gamma, float* __restrict__ outp) {
    __shared__ float red[P_DIM];
    int s = blockIdx.x, t = threadIdx.x;  // 128 threads
    const float* Ls = L + (long long)s * PP;
    const float* Lks = Lk + (long long)s * PP;
    const float* As = A + (long long)s * PP;
    const float* MAs = MA + (long long)s * PP;
    float delta = __expf(log_delta[0]), gam = __expf(log_gamma[0]);
    float prior_nu = delta + (float)P_DIM + 1.0f;
    float post_nu = prior_nu + gam;

    float logdet_L = bsum(__logf(Ls[t * P_DIM + t]), red);
    float logdet_A = bsum(__logf(As[t * P_DIM + t]), red);
    float logdet_Lk = bsum(__logf(Lks[t * P_DIM + t]), red);
    float a1 = 0.0f, a2 = 0.0f;
    for (int r = 0; r < P_DIM; ++r) {
        float av = As[r * P_DIM + t];
        a1 += av * av;
        float mv = MAs[r * P_DIM + t];
        a2 += mv * mv;
    }
    float trQ = bsum(a1, red);
    float trP = bsum(a2, red);
    float mvl_post = bsum(lgammaf(0.5f * post_nu - 0.5f * (float)t), red);
    float mvl_prior = bsum(lgammaf(0.5f * prior_nu - 0.5f * (float)t), red);

    if (t == 0) {
        const float LOGPI = 1.14472988584940017f;
        const float LOG2F = 0.69314718055994531f;
        float mvc = 0.25f * (float)P_DIM * (float)(P_DIM - 1) * LOGPI;
        mvl_post += mvc;
        mvl_prior += mvc;
        float logdet_Gii = 2.0f * (logdet_L - logdet_A);
        float logQ = -(post_nu + (float)P_DIM + 1.0f) * 0.5f * logdet_Gii +
                     post_nu * logdet_L - 0.5f * trQ -
                     post_nu * ((float)P_DIM * 0.5f) * LOG2F - mvl_post;
        float logP = -(prior_nu + (float)P_DIM + 1.0f) * 0.5f * logdet_Gii +
                     prior_nu * logdet_Lk - 0.5f * trP -
                     prior_nu * ((float)P_DIM * 0.5f) * LOG2F - mvl_prior;
        outp[s] = logP - logQ;
    }
}

// ---------------- host-side orchestration ----------------
extern "C" void kernel_launch(void* const* d_in, const int* in_sizes, int n_in,
                              void* d_out, int out_size, void* d_ws, size_t ws_size,
                              hipStream_t stream) {
    (void)in_sizes; (void)n_in; (void)out_size; (void)ws_size;
    const float* Kii = (const float*)d_in[0];
    const float* Kit = (const float*)d_in[1];
    const float* Ktt = (const float*)d_in[2];
    const float* V = (const float*)d_in[3];
    const float* log_delta = (const float*)d_in[4];
    const float* log_diag = (const float*)d_in[5];
    const float* log_gamma = (const float*)d_in[6];

    float* out = (float*)d_out;
    float* outGii = out;                                   // [S,P,P]
    float* outGit = out + (long long)S_DIM * PP;           // [S,P,PT]
    float* outGtt = outGit + (long long)S_DIM * PPT;       // [S,PT]
    float* outLogpq = outGtt + (long long)S_DIM * PT_DIM;  // [S]

    // workspace layout (floats)
    float* ws = (float*)d_ws;
    float* W = ws;                              // 16384
    float* L = W + PP;                          // S*PP : psi -> chol L
    float* Lk = L + (long long)S_DIM * PP;      // S*PP : dKii -> chol Lk
    float* Ab = Lk + (long long)S_DIM * PP;     // S*PP : Bartlett A
    float* Ainv = Ab + (long long)S_DIM * PP;
    float* Linv = Ainv + (long long)S_DIM * PP;
    float* Lkinv = Linv + (long long)S_DIM * PP;
    float* Bt = Lkinv + (long long)S_DIM * PP;
    float* Mm = Bt + (long long)S_DIM * PP;
    float* MAb = Mm + (long long)S_DIM * PP;
    float* gttiB = MAb + (long long)S_DIM * PP;         // S*PT
    float* Big1 = gttiB + (long long)S_DIM * PT_DIM;    // S*PPT : T1 -> noise
    float* Big2 = Big1 + (long long)S_DIM * PPT;        // S*PPT : invKii_kit -> X

    const size_t ldsChol = (size_t)P_DIM * LDP * sizeof(float);
    const size_t ldsTri = 2 * ldsChol;
    const size_t ldsPanel = (size_t)PP * sizeof(float);  // 64KB A panel for TDM

    // 1. W = diag * V V^T / P ; psi/dKii
    k_vdv<<<P_DIM, P_DIM, 0, stream>>>(V, log_diag, W);
    k_prep<<<(S_DIM * PP) / 256, 256, 0, stream>>>(Kii, W, log_delta, L, Lk);

    // 2. Choleskys (in place)
    k_chol<<<S_DIM, P_DIM, ldsChol, stream>>>(L);
    k_chol<<<S_DIM, P_DIM, ldsChol, stream>>>(Lk);

    // 3. Bartlett sample
    k_bartlett<<<(S_DIM * PP) / 256, 256, 0, stream>>>(Ab, log_delta, log_gamma);

    // 4. Triangular inverses (turn all solves into WMMA GEMMs)
    k_trinv<<<S_DIM, P_DIM, ldsTri, stream>>>(Ab, Ainv);
    k_trinv<<<S_DIM, P_DIM, ldsTri, stream>>>(L, Linv);
    k_trinv<<<S_DIM, P_DIM, ldsTri, stream>>>(Lk, Lkinv);

    // 5. Bt = A^{-1} L^T  (B(k,n) = L[n][k])
    k_gemm_small<<<dim3(8, 8, S_DIM), 32, 0, stream>>>(
        Ainv, L, Bt, PP, PP, PP, P_DIM, 1, 1, P_DIM, P_DIM);
    // 6. Gii = Bt^T Bt  -> output
    k_gemm_small<<<dim3(8, 8, S_DIM), 32, 0, stream>>>(
        Bt, Bt, outGii, PP, PP, PP, 1, P_DIM, P_DIM, 1, P_DIM);
    // 7. M = L^{-1} Lk
    k_gemm_small<<<dim3(8, 8, S_DIM), 32, 0, stream>>>(
        Linv, Lk, Mm, PP, PP, PP, P_DIM, 1, P_DIM, 1, P_DIM);
    // 8. MA = M^T A   (for trP)
    k_gemm_small<<<dim3(8, 8, S_DIM), 32, 0, stream>>>(
        Mm, Ab, MAb, PP, PP, PP, 1, P_DIM, P_DIM, 1, P_DIM);

    // 9. T1 = Lk^{-1} (delta*Kit)        (A panel via TDM->LDS)
    k_gemm_big<false><<<dim3(PT_DIM / 16, 1, S_DIM), 256, ldsPanel, stream>>>(
        Lkinv, Kit, Big1, PP, PPT, PPT, P_DIM, 1, PT_DIM, 1, PT_DIM, log_delta, 0u);
    // 10. invKii_kit = Lk^{-T} T1
    k_gemm_big<false><<<dim3(PT_DIM / 16, 1, S_DIM), 256, ldsPanel, stream>>>(
        Lkinv, Big1, Big2, PP, PPT, PPT, 1, P_DIM, PT_DIM, 1, PT_DIM, nullptr, 0u);
    // 11. gtti (must read Big2 before it becomes X)
    k_gtti<<<(S_DIM * PT_DIM) / 256, 256, 0, stream>>>(Kit, Ktt, Big2, log_delta, gttiB);
    // 12. noise = Lk^{-T} eps, eps generated in the loader (overwrites Big1)
    k_gemm_big<true><<<dim3(PT_DIM / 16, 1, S_DIM), 256, ldsPanel, stream>>>(
        Lkinv, Kit, Big1, PP, 0, PPT, 1, P_DIM, 0, 0, PT_DIM, nullptr, 0xA511E9B3u);
    // 13. X = invKii_kit + noise * sqrt(gtti)   (in place on Big2)
    k_form_X<<<(S_DIM * PPT) / 256, 256, 0, stream>>>(Big2, Big1, gttiB);
    // 14. git = Gii @ X  -> output
    k_gemm_big<false><<<dim3(PT_DIM / 16, 1, S_DIM), 256, ldsPanel, stream>>>(
        outGii, Big2, outGit, PP, PPT, PPT, P_DIM, 1, PT_DIM, 1, PT_DIM, nullptr, 0u);
    // 15. gtt -> output
    k_gtt<<<(S_DIM * PT_DIM) / 256, 256, 0, stream>>>(outGit, Big2, gttiB, outGtt);
    // 16. logpq -> output
    k_logpq<<<S_DIM, P_DIM, 0, stream>>>(L, Lk, Ab, MAb, log_delta, log_gamma, outLogpq);
}
